// E3_TransformerLayer_multi_27754078667290
// MI455X (gfx1250) — matrix operand order repaired and checked by reading, not compiled, not run on hardware
//
#include <hip/hip_runtime.h>
#include <hip/hip_fp16.h>

typedef __attribute__((ext_vector_type(16))) _Float16 v16h;
typedef __attribute__((ext_vector_type(8)))  float    v8f;

#define WMMA32(a,b,c) __builtin_amdgcn_wmma_f32_16x16x32_f16(false,(a),false,(b),(short)0,(c),false,false)

// ---- constants --------------------------------------------------------------
#define SQRT3      1.7320508f
#define SQRT5      2.2360680f
#define SQRT15     3.8729833f
#define INV_SQRT3  0.57735027f
#define INV_SQRT5  0.44721360f
#define INV_SQRT8  0.35355339f
#define INV_SQRT32 0.17677670f
#define INV_SQRT64 0.125f
#define INV_SQRT96 0.10206207f
#define INV_SQRT128 0.08838835f
#define CST_SILU   1.6790f          // 1/sqrt(E[silu(x)^2]), x~N(0,1)
#define EMB_SCALE  2.5253813f       // sqrt(8)/1.12
#define RB_STEP    0.555555556f     // R_MAX/(NB+1) = 5/9
#define INV_AVG    0.0625f          // 1/AVG_NEIGH

// Wigner 3j constants (1x1->2 family), unit-Frobenius normalization
#define W3A 0.31622777f
#define W3B 0.18257419f
#define W3T 0.36514837f
// (2,2,2) sparse table values
#define W3G1 0.23904572f
#define W3G2 0.11952286f
#define W3G3 0.35856858f

__device__ __constant__ int   C222I[19] = {0,3,0,1,1,3, 0,0,2, 3,3,2, 1,1,2, 4,4,2, 2};
__device__ __constant__ int   C222J[19] = {3,0,1,0,3,1, 0,2,0, 3,2,3, 1,2,1, 4,2,4, 2};
__device__ __constant__ int   C222K[19] = {1,1,3,3,0,0, 2,0,0, 2,3,3, 2,1,1, 2,4,4, 2};
__device__ __constant__ float C222V[19] = { W3G1, W3G1, W3G1, W3G1, W3G1, W3G1,
                                            W3G2, W3G2, W3G2,  W3G2, W3G2, W3G2,
                                           -W3G2,-W3G2,-W3G2, -W3G2,-W3G2,-W3G2, -W3G3};

// ---- CDNA5 WMMA fragment index helpers (16x16x32 f16, wave32) ---------------
// A (16x32, 16-bit): lanes 0-15: K in {0..7,16..23}; lanes 16-31: K in {8..15,24..31}
__device__ __forceinline__ int a_k(int lane, int h) {
  const int v = h >> 1, lo = h & 1;
  const int kb = (lane & 16) ? 8 : 0;
  return (v < 4) ? (kb + 2*v + lo) : (16 + kb + 2*(v - 4) + lo);
}
// B (32x16, 16-bit): lanes 0-15: K in {0..7,16..23}; lanes 16-31: K in {8..15,24..31}
__device__ __forceinline__ int b_k(int lane, int h) {
  const int v = h >> 1, lo = h & 1;
  const int kb = (lane & 16) ? 8 : 0;
  return (v < 4) ? (kb + 2*v + lo) : (16 + kb + 2*(v - 4) + lo);
}
// C/D (16x16 f32): element v -> row = v + (lane<16?0:8), col = lane&15
__device__ __forceinline__ int d_row(int lane, int v) {
  return v + ((lane & 16) ? 8 : 0);
}

__device__ __forceinline__ float silu_n(float x) {
  return x * (1.0f / (1.0f + __expf(-x))) * CST_SILU;
}

// ---- kernel 0: zero the node accumulator -----------------------------------
__global__ void k_zero(float* __restrict__ p, size_t n) {
  size_t i = (size_t)blockIdx.x * blockDim.x + threadIdx.x;
  if (i < n) p[i] = 0.0f;
}

// ---- kernel 1: node up-projection  f_in[N,288] -> xs[N,288] ----------------
// Templated on (OFF, D) so all addressing is compile-time constant.
template<int OFF, int D>
__device__ __forceinline__ void up_project(
    const float* __restrict__ f_in, const float* __restrict__ W,
    float* __restrict__ xs, int base, int arow, int lane, int col,
    bool full, int n_nodes)
{
  #pragma unroll
  for (int i = 0; i < D; ++i) {
    v16h a;
    const float* fr = f_in + (size_t)arow * 288 + OFF + i;
    #pragma unroll
    for (int h = 0; h < 16; ++h) a[h] = (_Float16)fr[a_k(lane, h) * D];
    #pragma unroll
    for (int t = 0; t < 2; ++t) {
      v16h b;
      #pragma unroll
      for (int h = 0; h < 16; ++h)
        b[h] = (_Float16)(W[b_k(lane, h) * 32 + t*16 + col] * INV_SQRT32);
      v8f c = {};
      c = WMMA32(a, b, c);
      if (full) {            // wave-uniform fast path: no per-element guards
        #pragma unroll
        for (int v = 0; v < 8; ++v)
          xs[(size_t)(base + d_row(lane, v)) * 288 + OFF + (t*16 + col) * D + i] = c[v];
      } else {
        #pragma unroll
        for (int v = 0; v < 8; ++v) {
          const int node = base + d_row(lane, v);
          if (node < n_nodes)
            xs[(size_t)node * 288 + OFF + (t*16 + col) * D + i] = c[v];
        }
      }
    }
  }
}

__global__ __launch_bounds__(128) void k_node_up(
    const float* __restrict__ f_in,
    const float* __restrict__ Wu0, const float* __restrict__ Wu1,
    const float* __restrict__ Wu2,
    float* __restrict__ xs, int n_nodes)
{
  const int wv   = (blockIdx.x * blockDim.x + threadIdx.x) >> 5;
  const int lane = threadIdx.x & 31;
  const int ntiles = (n_nodes + 15) >> 4;
  if (wv >= ntiles) return;
  const int base = wv << 4;
  const bool full = (base + 16 <= n_nodes);
  int arow = base + (lane & 15);
  if (arow > n_nodes - 1) arow = n_nodes - 1;
  const int col = lane & 15;

  up_project<0,   1>(f_in, Wu0, xs, base, arow, lane, col, full, n_nodes);
  up_project<32,  3>(f_in, Wu1, xs, base, arow, lane, col, full, n_nodes);
  up_project<128, 5>(f_in, Wu2, xs, base, arow, lane, col, full, n_nodes);
}

// ---- kernel 2: per-edge radial MLP (WMMA) + fused tensor product + scatter --
__global__ __launch_bounds__(64) void k_edge(
    const float* __restrict__ pos,    const int* __restrict__ batch,
    const int* __restrict__ esrc,     const int* __restrict__ edst,
    const float* __restrict__ eshift, const float* __restrict__ cell,
    const float* __restrict__ Wr0,    const float* __restrict__ Wr1,
    const float* __restrict__ Wr2,    const float* __restrict__ Wr3,
    const float* __restrict__ xs,     float* __restrict__ agg,
    int n_edges)
{
  __shared__ float emb_s[2][16][8];
  __shared__ float sh_s [2][16][8];
  __shared__ int   ed_s [2][16][2];
  __shared__ float h_s  [2][16][64];
  __shared__ float w_s  [2][16][352];

  const int wv   = threadIdx.x >> 5;
  const int lane = threadIdx.x & 31;
  const int ntiles = (n_edges + 15) >> 4;
  int tile = blockIdx.x * 2 + wv;
  const bool wvalid = (tile < ntiles);
  if (!wvalid) tile = ntiles - 1;       // keep lockstep for __syncthreads
  const int base = tile << 4;
  const int row = lane & 15;
  const int col = lane & 15;

  // --- phase 1: per-edge geometry, spherical harmonics, radial embedding ----
  if (lane < 16) {
    int e = base + lane; if (e > n_edges - 1) e = n_edges - 1;
    const int s = esrc[e], dd = edst[e];
    ed_s[wv][lane][0] = s; ed_s[wv][lane][1] = dd;
    const float* cb = cell + batch[s] * 9;
    const float sx = eshift[e*3+0], sy = eshift[e*3+1], sz = eshift[e*3+2];
    const float shx = sx*cb[0] + sy*cb[3] + sz*cb[6];
    const float shy = sx*cb[1] + sy*cb[4] + sz*cb[7];
    const float shz = sx*cb[2] + sy*cb[5] + sz*cb[8];
    const float vx = pos[dd*3+0] - pos[s*3+0] + shx;
    const float vy = pos[dd*3+1] - pos[s*3+1] + shy;
    const float vz = pos[dd*3+2] - pos[s*3+2] + shz;
    const float len = sqrtf(vx*vx + vy*vy + vz*vz);
    const float inv = 1.0f / fmaxf(len, 1e-12f);
    const float x = vx*inv, y = vy*inv, z = vz*inv;
    sh_s[wv][lane][0] = SQRT3 * x;
    sh_s[wv][lane][1] = SQRT3 * y;
    sh_s[wv][lane][2] = SQRT3 * z;
    sh_s[wv][lane][3] = SQRT15 * x * z;
    sh_s[wv][lane][4] = SQRT15 * x * y;
    sh_s[wv][lane][5] = SQRT5 * (y*y - 0.5f*(x*x + z*z));
    sh_s[wv][lane][6] = SQRT15 * y * z;
    sh_s[wv][lane][7] = 0.5f * SQRT15 * (z*z - x*x);
    #pragma unroll
    for (int j = 0; j < 8; ++j) {
      const float dlt = (len - RB_STEP * (float)(j + 1)) * (1.0f / RB_STEP);
      emb_s[wv][lane][j] = __expf(-dlt*dlt) * EMB_SCALE;
    }
  }
  __syncthreads();

  // --- layer 1: emb(16x8, K padded to 32) @ Wr0(8x64) -----------------------
  {
    v16h a;
    #pragma unroll
    for (int h = 0; h < 16; ++h) {
      const int k = a_k(lane, h);
      a[h] = (_Float16)((k < 8) ? emb_s[wv][row][k] : 0.0f);
    }
    v8f hc[4] = {};
    #pragma unroll
    for (int t = 0; t < 4; ++t) {
      v16h b;
      #pragma unroll
      for (int h = 0; h < 16; ++h) {
        const int k = b_k(lane, h);
        b[h] = (_Float16)((k < 8) ? Wr0[k*64 + t*16 + col] * INV_SQRT8 : 0.0f);
      }
      hc[t] = WMMA32(a, b, hc[t]);
    }
    #pragma unroll
    for (int t = 0; t < 4; ++t)
      #pragma unroll
      for (int v = 0; v < 8; ++v)
        h_s[wv][d_row(lane, v)][t*16 + col] = silu_n(hc[t][v]);
  }
  __syncthreads();

  // --- layers 2 & 3: h(16x64) @ W(64x64), silu ------------------------------
  for (int L = 0; L < 2; ++L) {
    const float* W = (L == 0) ? Wr1 : Wr2;
    v16h a0, a1;
    #pragma unroll
    for (int h = 0; h < 16; ++h) {
      a0[h] = (_Float16)h_s[wv][row][a_k(lane, h)];
      a1[h] = (_Float16)h_s[wv][row][32 + a_k(lane, h)];
    }
    v8f dc[4] = {};
    #pragma unroll
    for (int t = 0; t < 4; ++t) {
      v16h b0, b1;
      #pragma unroll
      for (int h = 0; h < 16; ++h) {
        b0[h] = (_Float16)(W[ b_k(lane, h)       * 64 + t*16 + col] * INV_SQRT64);
        b1[h] = (_Float16)(W[(32 + b_k(lane, h)) * 64 + t*16 + col] * INV_SQRT64);
      }
      dc[t] = WMMA32(a0, b0, dc[t]);
      dc[t] = WMMA32(a1, b1, dc[t]);
    }
    __syncthreads();  // all h_s reads done before overwrite
    #pragma unroll
    for (int t = 0; t < 4; ++t)
      #pragma unroll
      for (int v = 0; v < 8; ++v)
        h_s[wv][d_row(lane, v)][t*16 + col] = silu_n(dc[t][v]);
    __syncthreads();
  }

  // --- layer 4: h(16x64) @ Wr3(64x352) -> w_s (no activation) ---------------
  {
    v16h a0, a1;
    #pragma unroll
    for (int h = 0; h < 16; ++h) {
      a0[h] = (_Float16)h_s[wv][row][a_k(lane, h)];
      a1[h] = (_Float16)h_s[wv][row][32 + a_k(lane, h)];
    }
    for (int t = 0; t < 22; ++t) {
      v16h b0, b1;
      #pragma unroll
      for (int h = 0; h < 16; ++h) {
        b0[h] = (_Float16)(Wr3[ b_k(lane, h)       * 352 + t*16 + col] * INV_SQRT64);
        b1[h] = (_Float16)(Wr3[(32 + b_k(lane, h)) * 352 + t*16 + col] * INV_SQRT64);
      }
      v8f c = {};
      c = WMMA32(a0, b0, c);
      c = WMMA32(a1, b1, c);
      #pragma unroll
      for (int v = 0; v < 8; ++v)
        w_s[wv][d_row(lane, v)][t*16 + col] = c[v];
    }
  }
  __syncthreads();

  // --- fused tensor product + segment-sum scatter (lane = channel u) --------
  if (wvalid) {
    const int u = lane;
    int ecnt = n_edges - base; if (ecnt > 16) ecnt = 16;
    for (int ee = 0; ee < ecnt; ++ee) {
      const int s  = ed_s[wv][ee][0];
      const int dd = ed_s[wv][ee][1];
      const float* xp = xs + (size_t)s * 288;
      const float x0  = xp[u];
      const float x1x = xp[32 + u*3 + 0];
      const float x1y = xp[32 + u*3 + 1];
      const float x1z = xp[32 + u*3 + 2];
      float x2a[5];
      #pragma unroll
      for (int i2 = 0; i2 < 5; ++i2) x2a[i2] = xp[128 + u*5 + i2];
      const float s1x = sh_s[wv][ee][0], s1y = sh_s[wv][ee][1], s1z = sh_s[wv][ee][2];
      float ta[5];
      #pragma unroll
      for (int j2 = 0; j2 < 5; ++j2) ta[j2] = sh_s[wv][ee][3 + j2];
      const float* wsp = &w_s[wv][ee][0];
      float wk[11];
      #pragma unroll
      for (int p = 0; p < 11; ++p) wk[p] = wsp[p*32 + u];

      // l3 = 0 paths: (0,0,0), (1,1,0), (2,2,0)
      const float o00 = x0 * wk[0];
      const float o01 = INV_SQRT3 * (x1x*s1x + x1y*s1y + x1z*s1z) * wk[1];
      const float o02 = INV_SQRT5 *
          (x2a[0]*ta[0] + x2a[1]*ta[1] + x2a[2]*ta[2] + x2a[3]*ta[3] + x2a[4]*ta[4]) * wk[2];

      float o1v[4][3], o2v[4][5];
      // (0,1,1)
      o1v[0][0] = x0 * s1x * wk[3];
      o1v[0][1] = x0 * s1y * wk[3];
      o1v[0][2] = x0 * s1z * wk[3];
      // (1,0,1)
      o1v[1][0] = x1x * wk[4];
      o1v[1][1] = x1y * wk[4];
      o1v[1][2] = x1z * wk[4];
      // (1,2,1)
      {
        const float f = SQRT3 * wk[5];
        o1v[2][0] = f * (W3A*(x1y*ta[1] + x1z*ta[0]) - x1x*(W3B*ta[2] + W3A*ta[4]));
        o1v[2][1] = f * (W3A*(x1x*ta[1] + x1z*ta[3]) + W3T*x1y*ta[2]);
        o1v[2][2] = f * (W3A*(x1x*ta[0] + x1y*ta[3]) + x1z*(W3A*ta[4] - W3B*ta[2]));
      }
      // (2,1,1)
      {
        const float f = SQRT3 * wk[6];
        o1v[3][0] = f * (W3A*(s1y*x2a[1] + s1z*x2a[0]) - s1x*(W3B*x2a[2] + W3A*x2a[4]));
        o1v[3][1] = f * (W3A*(s1x*x2a[1] + s1z*x2a[3]) + W3T*s1y*x2a[2]);
        o1v[3][2] = f * (W3A*(s1x*x2a[0] + s1y*x2a[3]) + s1z*(W3A*x2a[4] - W3B*x2a[2]));
      }
      // (0,2,2)
      #pragma unroll
      for (int c2 = 0; c2 < 5; ++c2) o2v[0][c2] = x0 * ta[c2] * wk[7];
      // (1,1,2)
      {
        const float f = SQRT5 * wk[8];
        o2v[1][0] = f * W3A * (x1x*s1z + x1z*s1x);
        o2v[1][1] = f * W3A * (x1x*s1y + x1y*s1x);
        o2v[1][2] = f * (W3T*x1y*s1y - W3B*(x1x*s1x + x1z*s1z));
        o2v[1][3] = f * W3A * (x1y*s1z + x1z*s1y);
        o2v[1][4] = f * W3A * (x1z*s1z - x1x*s1x);
      }
      // (2,0,2)
      #pragma unroll
      for (int c2 = 0; c2 < 5; ++c2) o2v[2][c2] = x2a[c2] * wk[9];
      // (2,2,2)
      {
        const float f = SQRT5 * wk[10];
        float acc5[5] = {0.f, 0.f, 0.f, 0.f, 0.f};
        #pragma unroll
        for (int n2 = 0; n2 < 19; ++n2)
          acc5[C222K[n2]] += C222V[n2] * x2a[C222I[n2]] * ta[C222J[n2]];
        #pragma unroll
        for (int c2 = 0; c2 < 5; ++c2) o2v[3][c2] = f * acc5[c2];
      }

      float* ap = agg + (size_t)dd * 1120;
      atomicAdd(ap +      u, o00);
      atomicAdd(ap + 32 + u, o01);
      atomicAdd(ap + 64 + u, o02);
      #pragma unroll
      for (int p = 0; p < 4; ++p)
        #pragma unroll
        for (int c2 = 0; c2 < 3; ++c2)
          atomicAdd(ap + 96 + (p*32 + u)*3 + c2, o1v[p][c2]);
      #pragma unroll
      for (int p = 0; p < 4; ++p)
        #pragma unroll
        for (int c2 = 0; c2 < 5; ++c2)
          atomicAdd(ap + 480 + (p*32 + u)*5 + c2, o2v[p][c2]);
    }
  }
}

// ---- kernel 3: per-node output projections ---------------------------------
// Templated GEMM piece: K = KS*32, A from agg at (aoff + k*AD + i), out col stride AD.
template<int KS, int AOFF, int AD, int OOFF>
__device__ __forceinline__ void out_project(
    const float* __restrict__ ag, const float* __restrict__ W, float scaleB,
    float* __restrict__ out, int base, int lane, int col, int i,
    bool full, int n_nodes)
{
  v16h a[KS];
  #pragma unroll
  for (int ks = 0; ks < KS; ++ks)
    #pragma unroll
    for (int h = 0; h < 16; ++h)
      a[ks][h] = (_Float16)ag[AOFF + (ks*32 + a_k(lane, h)) * AD + i];
  #pragma unroll
  for (int t = 0; t < 2; ++t) {
    v8f c = {};
    #pragma unroll
    for (int ks = 0; ks < KS; ++ks) {
      v16h b;
      #pragma unroll
      for (int h = 0; h < 16; ++h)
        b[h] = (_Float16)(W[(ks*32 + b_k(lane, h))*32 + t*16 + col] * scaleB);
      c = WMMA32(a[ks], b, c);
    }
    if (full) {
      #pragma unroll
      for (int v = 0; v < 8; ++v)
        out[(size_t)(base + d_row(lane, v)) * 288 + OOFF + (t*16 + col) * AD + i]
            = c[v] * INV_AVG;
    } else {
      #pragma unroll
      for (int v = 0; v < 8; ++v) {
        const int node = base + d_row(lane, v);
        if (node < n_nodes)
          out[(size_t)node * 288 + OOFF + (t*16 + col) * AD + i] = c[v] * INV_AVG;
      }
    }
  }
}

__global__ __launch_bounds__(128) void k_out(
    const float* __restrict__ agg,
    const float* __restrict__ Wo0, const float* __restrict__ Wo1,
    const float* __restrict__ Wo2,
    float* __restrict__ out, int n_nodes)
{
  const int wv   = (blockIdx.x * blockDim.x + threadIdx.x) >> 5;
  const int lane = threadIdx.x & 31;
  const int ntiles = (n_nodes + 15) >> 4;
  if (wv >= ntiles) return;
  const int base = wv << 4;
  const bool full = (base + 16 <= n_nodes);
  int arow = base + (lane & 15);
  if (arow > n_nodes - 1) arow = n_nodes - 1;
  const float* ag = agg + (size_t)arow * 1120;
  const int col = lane & 15;

  // l = 0 : K=96, out cols 0..31
  out_project<3, 0, 1, 0>(ag, Wo0, INV_SQRT96, out, base, lane, col, 0, full, n_nodes);
  // l = 1 : K=128 per component i, out cols 32 + n*3 + i
  #pragma unroll
  for (int i = 0; i < 3; ++i)
    out_project<4, 96, 3, 32>(ag, Wo1, INV_SQRT128, out, base, lane, col, i, full, n_nodes);
  // l = 2 : K=128 per component i, out cols 128 + n*5 + i
  #pragma unroll
  for (int i = 0; i < 5; ++i)
    out_project<4, 480, 5, 128>(ag, Wo2, INV_SQRT128, out, base, lane, col, i, full, n_nodes);
}

// ---- launcher ---------------------------------------------------------------
extern "C" void kernel_launch(void* const* d_in, const int* in_sizes, int n_in,
                              void* d_out, int out_size, void* d_ws, size_t ws_size,
                              hipStream_t stream)
{
  (void)n_in; (void)out_size; (void)ws_size;
  const float* f_in   = (const float*)d_in[0];
  const float* pos    = (const float*)d_in[1];
  const int*   batch  = (const int*)d_in[2];
  const int*   esrc   = (const int*)d_in[3];
  const int*   edst   = (const int*)d_in[4];
  const float* eshift = (const float*)d_in[5];
  const float* cell   = (const float*)d_in[6];
  // d_in[7] = num_nodes scalar (device); use host-side in_sizes instead
  const float* Wu0 = (const float*)d_in[8];
  const float* Wu1 = (const float*)d_in[9];
  const float* Wu2 = (const float*)d_in[10];
  const float* Wr0 = (const float*)d_in[11];
  const float* Wr1 = (const float*)d_in[12];
  const float* Wr2 = (const float*)d_in[13];
  const float* Wr3 = (const float*)d_in[14];
  const float* Wo0 = (const float*)d_in[15];
  const float* Wo1 = (const float*)d_in[16];
  const float* Wo2 = (const float*)d_in[17];

  const int n_nodes = in_sizes[1] / 3;   // pos is (N,3)
  const int n_edges = in_sizes[3];       // edge_src is (E,)

  float* xs = (float*)d_ws;                                        // N x 288
  size_t xs_bytes = (((size_t)n_nodes * 288 * sizeof(float)) + 255) & ~(size_t)255;
  float* agg = (float*)((char*)d_ws + xs_bytes);                   // N x 1120

  const size_t agg_elems = (size_t)n_nodes * 1120;
  k_zero<<<(unsigned)((agg_elems + 255) / 256), 256, 0, stream>>>(agg, agg_elems);

  const int ntilesN = (n_nodes + 15) / 16;
  k_node_up<<<(ntilesN + 3) / 4, 128, 0, stream>>>(f_in, Wu0, Wu1, Wu2, xs, n_nodes);

  const int ntilesE = (n_edges + 15) / 16;
  k_edge<<<(ntilesE + 1) / 2, 64, 0, stream>>>(pos, batch, esrc, edst, eshift, cell,
                                               Wr0, Wr1, Wr2, Wr3, xs, agg, n_edges);

  k_out<<<(ntilesN + 3) / 4, 128, 0, stream>>>(agg, Wo0, Wo1, Wo2, (float*)d_out, n_nodes);
}
